// GRUDecoderModel_20521353740536
// MI455X (gfx1250) — matrix operand order repaired and checked by reading, not compiled, not run on hardware
//
#include <hip/hip_runtime.h>

// ---------------------------------------------------------------------------
// GRU decoder on MI455X (gfx1250): bf16 WMMA GEMMs with fp32 accumulation.
// Per step: fused (gi,gh,gates,h-update) kernel + fused (out,relu,bf16) kernel.
// Heads: WMMA GEMM to 128-dim hidden, tiny 3/6-dim projection from LDS.
// ---------------------------------------------------------------------------

typedef __attribute__((ext_vector_type(16))) __bf16 v16bf;
typedef __attribute__((ext_vector_type(8)))  __bf16 v8bf;
typedef __attribute__((ext_vector_type(8)))  float  v8f;

#define GRU_B   256
#define GRU_H   1024
#define GRU_I   512
#define SEQLEN  128
#define HDIM    128

__device__ __forceinline__ v8f wmma_bf16(v16bf a, v16bf b, v8f c) {
  // (neg_a, A, neg_b, B, c_mod, C, reuse_a, reuse_b)
  return __builtin_amdgcn_wmma_f32_16x16x32_bf16(false, a, false, b, (short)0, c,
                                                 false, false);
}

// A-matrix 16x32 bf16 fragment (ISA 7.12.2 16-bit A layout).
// lane L (0..15): row m0+L, K = {k0+e, k0+16+e : e=0..7}
// lane L (16..31): row m0+(L-15..): same row set, K groups shifted by 8.
__device__ __forceinline__ v16bf load_a16x32(const __bf16* __restrict__ base,
                                             int m0, int ld, int k0, int lane) {
  int row = m0 + (lane & 15);
  int kb  = (lane & 16) ? 8 : 0;
  const __bf16* p = base + (size_t)row * ld + k0 + kb;
  v8bf lo = *(const v8bf*)(p);        // K = k0+kb .. k0+kb+7
  v8bf hi = *(const v8bf*)(p + 16);   // K = k0+16+kb .. +7
  return __builtin_shufflevector(lo, hi, 0, 1, 2, 3, 4, 5, 6, 7,
                                 8, 9, 10, 11, 12, 13, 14, 15);
}

// B-matrix 32x16 bf16 fragment where B = W^T and W is [N,K] row-major:
// lane n holds output column n0+(n&15); K = {kb+e, 16+kb+e : e=0..7},
// kb = (n<16?0:8), from one W row (two b128 loads).
__device__ __forceinline__ v16bf load_bT32x16(const __bf16* __restrict__ w,
                                              int n0, int ld, int k0, int lane) {
  int row  = n0 + (lane & 15);
  int koff = (lane & 16) ? 8 : 0;
  const __bf16* p = w + (size_t)row * ld + k0 + koff;
  v8bf lo = *(const v8bf*)(p);
  v8bf hi = *(const v8bf*)(p + 16);
  return __builtin_shufflevector(lo, hi, 0, 1, 2, 3, 4, 5, 6, 7,
                                 8, 9, 10, 11, 12, 13, 14, 15);
}

__device__ __forceinline__ float sigmoid_f(float x) {
  return 1.0f / (1.0f + __expf(-x));
}
__device__ __forceinline__ float tanh_f(float x) {
  float ax = fabsf(x);
  float e  = __expf(-2.0f * ax);
  float t  = (1.0f - e) / (1.0f + e);
  return copysignf(t, x);
}

// --------------------------- setup kernels ---------------------------------
__global__ void cvt_f32_bf16(const float* __restrict__ src,
                             __bf16* __restrict__ dst, int n) {
  int i = blockIdx.x * blockDim.x + threadIdx.x;
  if (i < n) dst[i] = (__bf16)src[i];
}

__global__ void init_h_kernel(const float* __restrict__ eh,
                              float* __restrict__ hf,
                              __bf16* __restrict__ hb, int n) {
  int i = blockIdx.x * blockDim.x + threadIdx.x;
  if (i < n) {
    float v = eh[i];
    hf[i] = v;
    hb[i] = (__bf16)v;
  }
}

__global__ void zero_bf16_kernel(__bf16* __restrict__ p, int n) {
  int i = blockIdx.x * blockDim.x + threadIdx.x;
  if (i < n) p[i] = (__bf16)0.0f;
}

// ------------------- fused GRU gate + hidden-update kernel ------------------
// grid.x tiles H (64 cols/block), grid.y tiles B (32 rows/block).
// 8 waves: wm = wid&1 (16-row subtile), wn = wid>>1 (16-col subtile).
// Six fp32 accumulators per wave: {ir,iz,in} from Xr@W_ih^T, {hr,hz,hn} from
// h@W_hh^T; gate math fused in epilogue, writes h (f32) and h (bf16, ping-pong).
__global__ __launch_bounds__(256) void gru_gates_kernel(
    const __bf16* __restrict__ Xr,    // [256 x 512] relu(x) bf16
    const __bf16* __restrict__ Hin,   // [256 x 1024] bf16 (prev h)
    const __bf16* __restrict__ Wih,   // [3072 x 512] bf16
    const __bf16* __restrict__ Whh,   // [3072 x 1024] bf16
    const float* __restrict__ b_ih,   // [3072]
    const float* __restrict__ b_hh,   // [3072]
    float* __restrict__ Hf,           // [256 x 1024] f32 in/out (in-place)
    __bf16* __restrict__ Hout) {      // [256 x 1024] bf16 out (other buffer)
  const int tid  = threadIdx.x;
  const int wid  = tid >> 5;
  const int lane = tid & 31;
  const int m0 = blockIdx.y * 32 + (wid & 1) * 16;   // batch-row tile base
  const int n0 = blockIdx.x * 64 + (wid >> 1) * 16;  // hidden-col tile base

  v8f acc[6] = {};  // [0..2]=gi r,z,n  [3..5]=gh r,z,n

  // gi: K over input size 512
  for (int k = 0; k < GRU_I; k += 32) {
    v16bf a = load_a16x32(Xr, m0, GRU_I, k, lane);
#pragma unroll
    for (int g = 0; g < 3; ++g) {
      v16bf b = load_bT32x16(Wih, g * GRU_H + n0, GRU_I, k, lane);
      acc[g] = wmma_bf16(a, b, acc[g]);
    }
  }
  // gh: K over hidden size 1024
  for (int k = 0; k < GRU_H; k += 32) {
    v16bf a = load_a16x32(Hin, m0, GRU_H, k, lane);
#pragma unroll
    for (int g = 0; g < 3; ++g) {
      v16bf b = load_bT32x16(Whh, g * GRU_H + n0, GRU_H, k, lane);
      acc[3 + g] = wmma_bf16(a, b, acc[3 + g]);
    }
  }

  // Epilogue: C/D layout -> element i: row = m0 + i + (lane>=16 ? 8:0),
  //                                   col = n0 + (lane&15)
  const int col  = n0 + (lane & 15);
  const int rtop = m0 + ((lane & 16) ? 8 : 0);
  const float bir = b_ih[col],            bhr = b_hh[col];
  const float biz = b_ih[GRU_H + col],    bhz = b_hh[GRU_H + col];
  const float bin = b_ih[2 * GRU_H + col], bhn = b_hh[2 * GRU_H + col];
#pragma unroll
  for (int i = 0; i < 8; ++i) {
    const int row = rtop + i;
    float r = sigmoid_f(acc[0][i] + bir + acc[3][i] + bhr);
    float z = sigmoid_f(acc[1][i] + biz + acc[4][i] + bhz);
    float n = tanh_f(acc[2][i] + bin + r * (acc[5][i] + bhn));
    const size_t idx = (size_t)row * GRU_H + col;
    float hold = Hf[idx];
    float hnew = (1.0f - z) * n + z * hold;
    Hf[idx]   = hnew;
    Hout[idx] = (__bf16)hnew;
  }
}

// ------------------- out = h' @ W_out^T + b_out (fused relu/bf16) -----------
// grid.x tiles I (64 cols), grid.y tiles B (32 rows); wave tile 16x16.
__global__ __launch_bounds__(256) void gru_out_kernel(
    const __bf16* __restrict__ Hb,     // [256 x 1024] bf16 (new h)
    const __bf16* __restrict__ Wout,   // [512 x 1024] bf16
    const float* __restrict__ b_out,   // [512]
    __bf16* __restrict__ OutsRow,      // outs_bf16 + t*256*512
    __bf16* __restrict__ XrNext) {     // [256 x 512] relu(out) bf16
  const int tid  = threadIdx.x;
  const int wid  = tid >> 5;
  const int lane = tid & 31;
  const int m0 = blockIdx.y * 32 + (wid & 1) * 16;
  const int n0 = blockIdx.x * 64 + (wid >> 1) * 16;

  v8f acc = {};
  for (int k = 0; k < GRU_H; k += 32) {
    v16bf a = load_a16x32(Hb, m0, GRU_H, k, lane);
    v16bf b = load_bT32x16(Wout, n0, GRU_H, k, lane);
    acc = wmma_bf16(a, b, acc);
  }
  const int col  = n0 + (lane & 15);
  const int rtop = m0 + ((lane & 16) ? 8 : 0);
  const float bo = b_out[col];
#pragma unroll
  for (int i = 0; i < 8; ++i) {
    float v = acc[i] + bo;
    const size_t idx = (size_t)(rtop + i) * GRU_I + col;
    OutsRow[idx] = (__bf16)v;
    XrNext[idx]  = (__bf16)fmaxf(v, 0.0f);
  }
}

// ------------------------- MLP head (mean / cov) ----------------------------
// hidden = relu(outs @ W1^T + b1)  [32 x 128] per block via WMMA, staged in
// LDS, then tiny odim (3 or 6) projection. Row r of outs = (t*256 + b).
__global__ __launch_bounds__(256) void head_kernel(
    const __bf16* __restrict__ Outs,  // [32768 x 512] bf16
    const __bf16* __restrict__ W1,    // [128 x 512] bf16
    const float* __restrict__ b1,     // [128]
    const float* __restrict__ W2,     // [odim x 128] f32
    const float* __restrict__ b2,     // [odim]
    float* __restrict__ dst,          // output base (means or covs region)
    int odim) {
  __shared__ float hid[32 * HDIM];
  const int tid  = threadIdx.x;
  const int wid  = tid >> 5;
  const int lane = tid & 31;
  const int rowBase = blockIdx.x * 32;  // rows of Outs
  const int n0 = wid * 16;              // 8 waves cover 128 hidden cols

  v8f acc0 = {}, acc1 = {};
  for (int k = 0; k < GRU_I; k += 32) {
    v16bf b  = load_bT32x16(W1, n0, GRU_I, k, lane);
    v16bf a0 = load_a16x32(Outs, rowBase, GRU_I, k, lane);
    acc0 = wmma_bf16(a0, b, acc0);
    v16bf a1 = load_a16x32(Outs, rowBase + 16, GRU_I, k, lane);
    acc1 = wmma_bf16(a1, b, acc1);
  }
  const int col  = n0 + (lane & 15);
  const int rtop = (lane & 16) ? 8 : 0;
  const float bb = b1[col];
#pragma unroll
  for (int i = 0; i < 8; ++i) {
    hid[(rtop + i) * HDIM + col]      = fmaxf(acc0[i] + bb, 0.0f);
    hid[(rtop + i + 16) * HDIM + col] = fmaxf(acc1[i] + bb, 0.0f);
  }
  __syncthreads();

  for (int idx = tid; idx < 32 * odim; idx += 256) {
    const int lrow = idx / odim;
    const int od   = idx % odim;
    const float* hp = hid + lrow * HDIM;
    const float* w  = W2 + od * HDIM;
    float s = b2[od];
#pragma unroll 8
    for (int j = 0; j < HDIM; ++j) s += hp[j] * w[j];
    const int grow = rowBase + lrow;
    const int t = grow >> 8;     // row = t*256 + b
    const int b = grow & 255;
    dst[((size_t)b * SEQLEN + t) * odim + od] = s;
  }
}

// ------------------------------- launch -------------------------------------
extern "C" void kernel_launch(void* const* d_in, const int* in_sizes, int n_in,
                              void* d_out, int out_size, void* d_ws,
                              size_t ws_size, hipStream_t stream) {
  (void)in_sizes; (void)n_in; (void)out_size; (void)ws_size;
  const float* enc_hidden = (const float*)d_in[1];  // [1,256,1024]
  const float* W_ih  = (const float*)d_in[2];
  const float* W_hh  = (const float*)d_in[3];
  const float* b_ih  = (const float*)d_in[4];
  const float* b_hh  = (const float*)d_in[5];
  const float* W_out = (const float*)d_in[6];
  const float* b_out = (const float*)d_in[7];
  const float* Wm1   = (const float*)d_in[8];
  const float* bm1   = (const float*)d_in[9];
  const float* Wm2   = (const float*)d_in[10];
  const float* bm2   = (const float*)d_in[11];
  const float* Wc1   = (const float*)d_in[12];
  const float* bc1   = (const float*)d_in[13];
  const float* Wc2   = (const float*)d_in[14];
  const float* bc2   = (const float*)d_in[15];

  // Workspace layout (all 256B aligned by construction)
  char* ws = (char*)d_ws;
  size_t off = 0;
  __bf16* wih_b  = (__bf16*)(ws + off); off += (size_t)3 * GRU_H * GRU_I * 2;  // 3 MB
  __bf16* whh_b  = (__bf16*)(ws + off); off += (size_t)3 * GRU_H * GRU_H * 2;  // 6 MB
  __bf16* wout_b = (__bf16*)(ws + off); off += (size_t)GRU_I * GRU_H * 2;      // 1 MB
  __bf16* wm1_b  = (__bf16*)(ws + off); off += (size_t)HDIM * GRU_I * 2;
  __bf16* wc1_b  = (__bf16*)(ws + off); off += (size_t)HDIM * GRU_I * 2;
  float*  h_f32  = (float*)(ws + off);  off += (size_t)GRU_B * GRU_H * 4;      // 1 MB
  __bf16* h_b    = (__bf16*)(ws + off); off += (size_t)2 * GRU_B * GRU_H * 2;  // ping-pong
  __bf16* xr_b   = (__bf16*)(ws + off); off += (size_t)GRU_B * GRU_I * 2;
  __bf16* outs_b = (__bf16*)(ws + off); off += (size_t)SEQLEN * GRU_B * GRU_I * 2; // 32 MB

  const int T256 = 256;
  auto blocks = [](int n) { return (n + 255) / 256; };

  // One-time-per-call weight conversion (weights then live in L2 for 128 steps)
  cvt_f32_bf16<<<blocks(3 * GRU_H * GRU_I), T256, 0, stream>>>(W_ih, wih_b, 3 * GRU_H * GRU_I);
  cvt_f32_bf16<<<blocks(3 * GRU_H * GRU_H), T256, 0, stream>>>(W_hh, whh_b, 3 * GRU_H * GRU_H);
  cvt_f32_bf16<<<blocks(GRU_I * GRU_H), T256, 0, stream>>>(W_out, wout_b, GRU_I * GRU_H);
  cvt_f32_bf16<<<blocks(HDIM * GRU_I), T256, 0, stream>>>(Wm1, wm1_b, HDIM * GRU_I);
  cvt_f32_bf16<<<blocks(HDIM * GRU_I), T256, 0, stream>>>(Wc1, wc1_b, HDIM * GRU_I);
  init_h_kernel<<<blocks(GRU_B * GRU_H), T256, 0, stream>>>(enc_hidden, h_f32,
                                                            h_b, GRU_B * GRU_H);
  zero_bf16_kernel<<<blocks(GRU_B * GRU_I), T256, 0, stream>>>(xr_b, GRU_B * GRU_I);

  const size_t HN = (size_t)GRU_B * GRU_H;
  for (int t = 0; t < SEQLEN; ++t) {
    const __bf16* hin = h_b + (size_t)(t & 1) * HN;
    __bf16* hout      = h_b + (size_t)((t + 1) & 1) * HN;
    gru_gates_kernel<<<dim3(GRU_H / 64, GRU_B / 32), T256, 0, stream>>>(
        xr_b, hin, wih_b, whh_b, b_ih, b_hh, h_f32, hout);
    gru_out_kernel<<<dim3(GRU_I / 64, GRU_B / 32), T256, 0, stream>>>(
        hout, wout_b, b_out, outs_b + (size_t)t * GRU_B * GRU_I, xr_b);
  }

  // Heads over all S*B = 32768 rows
  float* means = (float*)d_out;
  float* covs  = means + (size_t)GRU_B * SEQLEN * 3;
  head_kernel<<<(SEQLEN * GRU_B) / 32, T256, 0, stream>>>(outs_b, wm1_b, bm1,
                                                          Wm2, bm2, means, 3);
  head_kernel<<<(SEQLEN * GRU_B) / 32, T256, 0, stream>>>(outs_b, wc1_b, bc1,
                                                          Wc2, bc2, covs, 6);
}
